// BiLSTM_CRF_14834817040755
// MI455X (gfx1250) — compile-verified, matrix operand order checked by reading, not run on hardware
//
#include <hip/hip_runtime.h>
#include <hip/hip_bf16.h>

typedef __attribute__((ext_vector_type(16))) __bf16 v16bf;
typedef __attribute__((ext_vector_type(8)))  __bf16 v8bf;
typedef __attribute__((ext_vector_type(8)))  float  v8f;

#define T_SEQ 4096
#define EMBD  256
#define HID   512
#define GATES (4 * HID)   // 2048
#define NTAGS 32
#define TAG_START 30
#define TAG_STOP  31
#define NEGV -10000.0f

// ---------------------------------------------------------------------------
// CDNA5 async-to-LDS helpers (ASYNCcnt-tracked DMA into LDS)
// ---------------------------------------------------------------------------
__device__ __forceinline__ void async_ld_b128(unsigned lds_off, const void* gaddr) {
    // per-lane: LDS[lds_off] = MEM[gaddr], 16 bytes, tracked by ASYNCcnt
    asm volatile("global_load_async_to_lds_b128 %0, %1, off"
                 :: "v"(lds_off), "v"(gaddr) : "memory");
}
__device__ __forceinline__ void wait_asynccnt0() {
    asm volatile("s_wait_asynccnt 0" ::: "memory");
}

// Stage a ROWSx32 bf16 tile (row stride = ldg elements in global, 32 in LDS)
// as ROWS*4 16-byte async transfers spread over 256 threads.
template <int ROWS>
__device__ __forceinline__ void stage_tile(__bf16* ldsTile, const __bf16* gbase,
                                           int ldg, int tid) {
    constexpr int SEGS = ROWS * 4;          // 16B segments
#pragma unroll 1
    for (int idx = tid; idx < SEGS; idx += 256) {
        const int row = idx >> 2;
        const int seg = idx & 3;
        const unsigned lds_off =
            (unsigned)(size_t)(ldsTile + row * 32) + seg * 16;
        const char* g = (const char*)(gbase + (size_t)row * ldg) + seg * 16;
        async_ld_b128(lds_off, g);
    }
}

// ---------------------------------------------------------------------------
// fp32 -> bf16 elementwise convert (WMMA operand staging)
// ---------------------------------------------------------------------------
__global__ void cvt_bf16_kernel(const float* __restrict__ src,
                                __bf16* __restrict__ dst, int n) {
    int i = blockIdx.x * blockDim.x + threadIdx.x;
    if (i < n) dst[i] = (__bf16)src[i];
}

// ---------------------------------------------------------------------------
// embedding gather: x[t][e] = emb[tokens[t]][e]  (bf16 out)
// ---------------------------------------------------------------------------
__global__ void gather_kernel(const int* __restrict__ tokens,
                              const float* __restrict__ emb,
                              __bf16* __restrict__ x) {
    int t = blockIdx.x;
    int e = threadIdx.x;                       // 256 threads = EMBD
    int tok = tokens[t];
    x[(size_t)t * EMBD + e] = (__bf16)emb[(size_t)tok * EMBD + e];
}

// ---------------------------------------------------------------------------
// LDS-tiled, async-double-buffered WMMA GEMM:
//   C(MxN) = A(MxK,bf16,rm) @ W(NxK,bf16,rm)^T + bias(N)
// Block = 256 threads (8 waves), macro tile BM x BN, K-slab of 32.
// Wave grid WAVESM x WAVESN; wave tile = (BM/WAVESM) x (BN/WAVESN) built from
// 16x16 WMMA subtiles -> A fragments reused across SN, B across SM.
// ---------------------------------------------------------------------------
template <int BM, int BN, int WAVESM, int WAVESN>
__global__ __launch_bounds__(256) void
wmma_gemm_lds_kernel(const __bf16* __restrict__ A, const __bf16* __restrict__ W,
                     const float* __restrict__ bias, float* __restrict__ C,
                     int M, int N, int K) {
    constexpr int WTM = BM / WAVESM;
    constexpr int WTN = BN / WAVESN;
    constexpr int SM  = WTM / 16;
    constexpr int SN  = WTN / 16;

    __shared__ __bf16 Atile[2][BM * 32];
    __shared__ __bf16 Btile[2][BN * 32];

    const int tid  = threadIdx.x;
    const int wave = tid >> 5;
    const int lane = tid & 31;
    const int half = lane >> 4;            // 0: lanes 0-15, 1: lanes 16-31
    const int l    = lane & 15;
    const int kb   = half * 8;             // K sub-offset for this half-wave

    const int wm = (wave / WAVESN) * WTM;  // wave tile origin in block tile
    const int wn = (wave % WAVESN) * WTN;

    const int m0 = blockIdx.y * BM;
    const int n0 = blockIdx.x * BN;

    const __bf16* Ablk = A + (size_t)m0 * K;
    const __bf16* Wblk = W + (size_t)n0 * K;

    v8f acc[SM][SN] = {};

    // prologue: stage first K-slab into buffer 0
    stage_tile<BM>(&Atile[0][0], Ablk, K, tid);
    stage_tile<BN>(&Btile[0][0], Wblk, K, tid);

    for (int kk = 0; kk < K; kk += 32) {
        const int buf = (kk >> 5) & 1;
        wait_asynccnt0();                  // this wave's async DMAs landed
        __syncthreads();                   // everyone's DMAs landed

        if (kk + 32 < K) {                 // kick off next slab into other buf
            stage_tile<BM>(&Atile[buf ^ 1][0], Ablk + kk + 32, K, tid);
            stage_tile<BN>(&Btile[buf ^ 1][0], Wblk + kk + 32, K, tid);
        }

        // fragment loads from LDS (ds_load_b128 pairs)
        v16bf afrag[SM], bfrag[SN];
#pragma unroll
        for (int s = 0; s < SM; ++s) {
            const __bf16* ap = &Atile[buf][(wm + s * 16 + l) * 32];
            v8bf lo = *(const v8bf*)(ap + kb);
            v8bf hi = *(const v8bf*)(ap + kb + 16);
            afrag[s] = __builtin_shufflevector(lo, hi, 0, 1, 2, 3, 4, 5, 6, 7,
                                               8, 9, 10, 11, 12, 13, 14, 15);
        }
#pragma unroll
        for (int s = 0; s < SN; ++s) {
            const __bf16* bp = &Btile[buf][(wn + s * 16 + l) * 32];
            v8bf lo = *(const v8bf*)(bp + kb);
            v8bf hi = *(const v8bf*)(bp + kb + 16);
            bfrag[s] = __builtin_shufflevector(lo, hi, 0, 1, 2, 3, 4, 5, 6, 7,
                                               8, 9, 10, 11, 12, 13, 14, 15);
        }
#pragma unroll
        for (int sm = 0; sm < SM; ++sm)
#pragma unroll
            for (int sn = 0; sn < SN; ++sn)
                acc[sm][sn] = __builtin_amdgcn_wmma_f32_16x16x32_bf16(
                    false, afrag[sm], false, bfrag[sn], (short)0, acc[sm][sn],
                    false, false);

        __syncthreads();                   // reads done before buf is re-staged
    }

    // epilogue: C/D layout -> VGPR r holds M = half*8 + r, N = l
#pragma unroll
    for (int sn = 0; sn < SN; ++sn) {
        const int n  = n0 + wn + sn * 16 + l;
        const float bn = bias ? bias[n] : 0.0f;
#pragma unroll
        for (int sm = 0; sm < SM; ++sm) {
#pragma unroll
            for (int r = 0; r < 8; ++r) {
                const int m = m0 + wm + sm * 16 + half * 8 + r;
                C[(size_t)m * N + n] = acc[sm][sn][r] + bn;
            }
        }
    }
}

// ---------------------------------------------------------------------------
// LSTM scan. blockIdx.x = direction (0 fwd, 1 bwd), 512 threads: thread j owns
// hidden unit j (gate rows j, j+H, j+2H, j+3H). h lives in LDS, c in a VGPR.
// ---------------------------------------------------------------------------
__global__ void lstm_scan_kernel(const float* __restrict__ Whh_f,
                                 const float* __restrict__ G_f,
                                 const float* __restrict__ h0_f,
                                 const float* __restrict__ c0_f,
                                 const float* __restrict__ Whh_b,
                                 const float* __restrict__ G_b,
                                 const float* __restrict__ h0_b,
                                 const float* __restrict__ c0_b,
                                 __bf16* __restrict__ Hcat) {
    __shared__ float hs[HID];
    const int dir = blockIdx.x;
    const int j = threadIdx.x;                 // 0..511

    const float* Whh = dir ? Whh_b : Whh_f;
    const float* G   = dir ? G_b   : G_f;
    const float* h0  = dir ? h0_b  : h0_f;
    const float* c0  = dir ? c0_b  : c0_f;

    const float* w0 = Whh + (size_t)(j)           * HID;  // i-gate row
    const float* w1 = Whh + (size_t)(j + HID)     * HID;  // f-gate row
    const float* w2 = Whh + (size_t)(j + 2 * HID) * HID;  // g-gate row
    const float* w3 = Whh + (size_t)(j + 3 * HID) * HID;  // o-gate row

    hs[j] = h0[j];
    float c = c0[j];
    __syncthreads();

    for (int t = 0; t < T_SEQ; ++t) {
        const float* g = G + (size_t)t * GATES;
        if (t + 1 < T_SEQ) __builtin_prefetch((const void*)(g + GATES + j), 0, 1);

        float ai = g[j];
        float af = g[j + HID];
        float ag = g[j + 2 * HID];
        float ao = g[j + 3 * HID];

        float di = 0.f, df = 0.f, dg = 0.f, dq = 0.f;
        for (int k = 0; k < HID; k += 4) {
            const float4 hv = *(const float4*)&hs[k];     // ds_load_b128 broadcast
            const float4 a0 = *(const float4*)&w0[k];
            const float4 a1 = *(const float4*)&w1[k];
            const float4 a2 = *(const float4*)&w2[k];
            const float4 a3 = *(const float4*)&w3[k];
            di += a0.x * hv.x + a0.y * hv.y + a0.z * hv.z + a0.w * hv.w;
            df += a1.x * hv.x + a1.y * hv.y + a1.z * hv.z + a1.w * hv.w;
            dg += a2.x * hv.x + a2.y * hv.y + a2.z * hv.z + a2.w * hv.w;
            dq += a3.x * hv.x + a3.y * hv.y + a3.z * hv.z + a3.w * hv.w;
        }
        ai += di; af += df; ag += dg; ao += dq;

        const float ig = 1.0f / (1.0f + __expf(-ai));
        const float fg = 1.0f / (1.0f + __expf(-af));
        const float gg = tanhf(ag);
        const float og = 1.0f / (1.0f + __expf(-ao));
        c = fg * c + ig * gg;
        const float h = og * tanhf(c);

        __syncthreads();          // all dots of step t done before overwrite
        hs[j] = h;
        const int tt = dir ? (T_SEQ - 1 - t) : t;
        Hcat[(size_t)tt * (2 * HID) + dir * HID + j] = (__bf16)h;
        __syncthreads();
    }
}

// ---------------------------------------------------------------------------
// Viterbi / CRF forward with backpointers. One wave32, lane = next-tag.
// out[0] = logsumexp score, out[1..T] = path (as float).
// ---------------------------------------------------------------------------
__global__ void viterbi_kernel(const float* __restrict__ feats,
                               const float* __restrict__ trans,
                               int* __restrict__ bp,
                               float* __restrict__ out) {
    __shared__ float tr[NTAGS * NTAGS];
    __shared__ float alpha[NTAGS];
    __shared__ float fin[NTAGS];
    const int lane = threadIdx.x;              // 32 threads

    for (int i = lane; i < NTAGS * NTAGS; i += NTAGS) tr[i] = trans[i];
    alpha[lane] = (lane == TAG_START) ? 0.0f : NEGV;
    __syncthreads();

    for (int t = 0; t < T_SEQ; ++t) {
        const float feat = feats[(size_t)t * NTAGS + lane];
        float m = -3.402823466e38f;
        int arg = 0;
#pragma unroll
        for (int p = 0; p < NTAGS; ++p) {
            const float v = alpha[p] + tr[p * NTAGS + lane];
            if (v > m) { m = v; arg = p; }     // first-max like jnp.argmax
        }
        float s = 0.0f;
#pragma unroll
        for (int p = 0; p < NTAGS; ++p) {
            const float v = alpha[p] + tr[p * NTAGS + lane];
            s += __expf(v - m);
        }
        bp[(size_t)t * NTAGS + lane] = arg;
        const float na = feat + m + __logf(s);
        __syncthreads();
        alpha[lane] = na;
        __syncthreads();
    }

    fin[lane] = alpha[lane] + tr[lane * NTAGS + TAG_STOP];
    __syncthreads();
    if (lane == 0) {
        float m = fin[0]; int best = 0;
        for (int p = 1; p < NTAGS; ++p)
            if (fin[p] > m) { m = fin[p]; best = p; }
        float s = 0.0f;
        for (int p = 0; p < NTAGS; ++p) s += __expf(fin[p] - m);
        out[0] = m + __logf(s);

        int cur = best;
        out[1 + (T_SEQ - 1)] = (float)cur;
        for (int t = T_SEQ - 1; t >= 1; --t) {
            cur = bp[(size_t)t * NTAGS + cur];
            out[1 + (t - 1)] = (float)cur;
        }
    }
}

// ---------------------------------------------------------------------------
// host launcher
// ---------------------------------------------------------------------------
extern "C" void kernel_launch(void* const* d_in, const int* in_sizes, int n_in,
                              void* d_out, int out_size, void* d_ws, size_t ws_size,
                              hipStream_t stream) {
    const int*   tokens = (const int*)  d_in[0];
    const float* emb    = (const float*)d_in[1];
    const float* W_ih_f = (const float*)d_in[2];
    const float* W_hh_f = (const float*)d_in[3];
    const float* b_f    = (const float*)d_in[4];
    const float* h0_f   = (const float*)d_in[5];
    const float* c0_f   = (const float*)d_in[6];
    const float* W_ih_b = (const float*)d_in[7];
    const float* W_hh_b = (const float*)d_in[8];
    const float* b_b    = (const float*)d_in[9];
    const float* h0_b   = (const float*)d_in[10];
    const float* c0_b   = (const float*)d_in[11];
    const float* W_out  = (const float*)d_in[12];
    const float* b_out  = (const float*)d_in[13];
    const float* trans  = (const float*)d_in[14];
    float* out = (float*)d_out;

    // workspace carve-up (256B aligned)
    char* ws = (char*)d_ws;
    size_t off = 0;
    auto carve = [&](size_t bytes) -> void* {
        void* p = ws + off;
        off = (off + bytes + 255) & ~(size_t)255;
        return p;
    };
    __bf16* x_bf    = (__bf16*)carve((size_t)T_SEQ * EMBD * 2);       //  2 MB
    __bf16* Wihf_bf = (__bf16*)carve((size_t)GATES * EMBD * 2);       //  1 MB
    __bf16* Wihb_bf = (__bf16*)carve((size_t)GATES * EMBD * 2);       //  1 MB
    __bf16* Wout_bf = (__bf16*)carve((size_t)NTAGS * 2 * HID * 2);    // 64 KB
    float*  Gf      = (float*) carve((size_t)T_SEQ * GATES * 4);      // 32 MB
    float*  Gb      = (float*) carve((size_t)T_SEQ * GATES * 4);      // 32 MB
    __bf16* Hcat    = (__bf16*)carve((size_t)T_SEQ * 2 * HID * 2);    //  8 MB
    float*  feats   = (float*) carve((size_t)T_SEQ * NTAGS * 4);      // .5 MB
    int*    bp      = (int*)   carve((size_t)T_SEQ * NTAGS * 4);      // .5 MB
    (void)ws_size; (void)in_sizes; (void)n_in; (void)out_size;

    // 1) stage bf16 operands
    {
        int n = GATES * EMBD;
        cvt_bf16_kernel<<<(n + 255) / 256, 256, 0, stream>>>(W_ih_f, Wihf_bf, n);
        cvt_bf16_kernel<<<(n + 255) / 256, 256, 0, stream>>>(W_ih_b, Wihb_bf, n);
        int m = NTAGS * 2 * HID;
        cvt_bf16_kernel<<<(m + 255) / 256, 256, 0, stream>>>(W_out, Wout_bf, m);
    }
    // 2) embedding gather
    gather_kernel<<<T_SEQ, EMBD, 0, stream>>>(tokens, emb, x_bf);

    // 3) input projections: G = X @ W_ih^T + b   (M=4096, N=2048, K=256)
    {
        dim3 grid(GATES / 128, T_SEQ / 64);    // (16, 64)
        wmma_gemm_lds_kernel<64, 128, 2, 4><<<grid, 256, 0, stream>>>(
            x_bf, Wihf_bf, b_f, Gf, T_SEQ, GATES, EMBD);
        wmma_gemm_lds_kernel<64, 128, 2, 4><<<grid, 256, 0, stream>>>(
            x_bf, Wihb_bf, b_b, Gb, T_SEQ, GATES, EMBD);
    }
    // 4) both LSTM scans (2 blocks, 512 threads each)
    lstm_scan_kernel<<<2, HID, 0, stream>>>(W_hh_f, Gf, h0_f, c0_f,
                                            W_hh_b, Gb, h0_b, c0_b, Hcat);
    // 5) tag projection: feats = Hcat @ W_out^T + b_out (M=4096, N=32, K=1024)
    {
        dim3 grid(NTAGS / 32, T_SEQ / 128);    // (1, 32)
        wmma_gemm_lds_kernel<128, 32, 4, 2><<<grid, 256, 0, stream>>>(
            Hcat, Wout_bf, b_out, feats, T_SEQ, NTAGS, 2 * HID);
    }
    // 6) CRF viterbi + backtrace
    viterbi_kernel<<<1, NTAGS, 0, stream>>>(feats, trans, bp, out);
}
